// RecGAT_51170240364905
// MI455X (gfx1250) — matrix operand order repaired and checked by reading, not compiled, not run on hardware
//
#include <hip/hip_runtime.h>
#include <hip/hip_bf16.h>
#include <math.h>

#define DD 128

typedef __bf16 bf16;
typedef bf16  v16bf __attribute__((ext_vector_type(16)));
typedef float v8f   __attribute__((ext_vector_type(8)));

__device__ __forceinline__ float4 ld4(const float* p) { return *(const float4*)p; }

// float atomic max via int-bit trick (valid with -inf init)
__device__ __forceinline__ void atomicMaxF(float* a, float v) {
    if (v >= 0.0f) atomicMax((int*)a,          __float_as_int(v));
    else           atomicMin((unsigned int*)a, __float_as_uint(v));
}

// ---------------------------------------------------------------------------
// GEMM: O_w[r][n] = sum_k X[r][k] * W_w[k][n] + b_w[n]  for w in {0,1,2}
// X read once per block (converted to bf16 in LDS), WMMA f32<-bf16.
// Block: 256 threads = 8 waves. Tile: 128 rows x 128 cols, K = 128.
// C is staged through LDS (overlapping the B buffer) for coalesced stores.
// ---------------------------------------------------------------------------
__global__ __launch_bounds__(256)
void gemm3_wmma(const float* __restrict__ X, int nrows,
                const float* __restrict__ W0, const float* __restrict__ b0, float* __restrict__ O0,
                const float* __restrict__ W1, const float* __restrict__ b1, float* __restrict__ O1,
                const float* __restrict__ W2, const float* __restrict__ b2, float* __restrict__ O2)
{
    __shared__ __align__(16) bf16 Asm[128][DD];                       // 32 KB, persistent
    __shared__ __align__(16) unsigned char BC[128 * DD * sizeof(float)]; // 64 KB: Bt / Csm overlap
    bf16  (*Bt )[DD] = (bf16 (*)[DD])BC;   // Bt[n][k] = W[k][n], bf16
    float (*Csm)[DD] = (float(*)[DD])BC;   // C staging, f32

    const int t       = threadIdx.x;
    const int rowBase = blockIdx.x * 128;

    // ---- load & convert A tile (clamp out-of-range rows)
    #pragma unroll
    for (int it = 0; it < 16; ++it) {
        int f  = t + it * 256;         // float4 index within 128x128 tile
        int r  = f >> 5;               // 32 float4 per row
        int c4 = (f & 31) << 2;
        int gr = rowBase + r; if (gr >= nrows) gr = nrows - 1;
        float4 v = ld4(X + (size_t)gr * DD + c4);
        Asm[r][c4 + 0] = (bf16)v.x; Asm[r][c4 + 1] = (bf16)v.y;
        Asm[r][c4 + 2] = (bf16)v.z; Asm[r][c4 + 3] = (bf16)v.w;
    }

    const int wave = t >> 5;
    const int lane = t & 31;
    const int lh   = lane & 15;   // 0..15
    const int hi   = lane >> 4;   // 0 or 1
    const bool fullTile = (rowBase + 128) <= nrows;

    const float* Ws[3] = { W0, W1, W2 };
    const float* bs[3] = { b0, b1, b2 };
    float*       Os[3] = { O0, O1, O2 };

    union AB { v16bf v; float4 f4[2]; };

    #pragma unroll
    for (int w = 0; w < 3; ++w) {
        __syncthreads();   // previous Csm reads done before overwriting with Bt
        // ---- load W, store transposed bf16 into Bt[n][k]
        const float* W = Ws[w];
        #pragma unroll
        for (int it = 0; it < 16; ++it) {
            int f  = t + it * 256;
            int k  = f >> 5;
            int c4 = (f & 31) << 2;
            float4 v = ld4(W + k * DD + c4);
            Bt[c4 + 0][k] = (bf16)v.x; Bt[c4 + 1][k] = (bf16)v.y;
            Bt[c4 + 2][k] = (bf16)v.z; Bt[c4 + 3][k] = (bf16)v.w;
        }
        __syncthreads();

        // ---- init 8 accumulator tiles (whole 16x128 strip per wave)
        const float* bias = bs[w];
        v8f acc[8];
        #pragma unroll
        for (int n = 0; n < 8; ++n) {
            const float bv = bias ? bias[(n << 4) + lh] : 0.0f;
            #pragma unroll
            for (int j = 0; j < 8; ++j) acc[n][j] = bv;
        }

        // ---- K loop: A frag loaded once per kk, reused for all 8 N tiles.
        // ISA 16-bit A 16x32 layout: lane<16 holds K=[kb..kb+7],[kb+16..kb+23];
        // lane>=16 holds K=[kb+8..kb+15],[kb+24..kb+31].
        // ISA 16-bit B 32x16 layout: same K order as A: lanes 0-15 hold K=[kb..kb+7],[kb+16..kb+23] (col=lane),
        // lanes 16-31 hold K=[kb+8..kb+15],[kb+24..kb+31] in Bt[col][.]
        const int arow = (wave << 4) + lh;
        #pragma unroll
        for (int kk = 0; kk < 4; ++kk) {
            AB a;
            int ka = kk * 32 + hi * 8;
            a.f4[0] = *(const float4*)&Asm[arow][ka];
            a.f4[1] = *(const float4*)&Asm[arow][ka + 16];
            int kb = kk * 32 + hi * 8;
            #pragma unroll
            for (int n = 0; n < 8; ++n) {
                AB b;
                const int col = (n << 4) + lh;
                b.f4[0] = *(const float4*)&Bt[col][kb];
                b.f4[1] = *(const float4*)&Bt[col][kb + 16];
                acc[n] = __builtin_amdgcn_wmma_f32_16x16x32_bf16(
                             false, a.v, false, b.v, (short)0, acc[n], false, false);
            }
        }

        __syncthreads();   // all waves done reading Bt -> safe to overwrite with Csm

        // ---- stage C to LDS (C layout: VGPR j -> row j + 8*hi, lane -> col)
        #pragma unroll
        for (int n = 0; n < 8; ++n) {
            const int col = (n << 4) + lh;
            const int rw  = (wave << 4) + (hi << 3);
            #pragma unroll
            for (int j = 0; j < 8; ++j) Csm[rw + j][col] = acc[n][j];
        }
        __syncthreads();

        // ---- coalesced global store of the 128x128 tile
        float* O = Os[w];
        if (fullTile) {
            #pragma unroll
            for (int it = 0; it < 16; ++it) {
                int f  = t + it * 256;
                int r  = f >> 5;
                int c4 = (f & 31) << 2;
                *(float4*)(O + (size_t)(rowBase + r) * DD + c4) = *(const float4*)&Csm[r][c4];
            }
        } else {
            #pragma unroll
            for (int it = 0; it < 16; ++it) {
                int f  = t + it * 256;
                int r  = f >> 5;
                int c4 = (f & 31) << 2;
                int gr = rowBase + r;
                if (gr < nrows)
                    *(float4*)(O + (size_t)gr * DD + c4) = *(const float4*)&Csm[r][c4];
            }
        }
    }
}

// ---------------------------------------------------------------------------
// Edge / segment kernels (memory-bound phase)
// ---------------------------------------------------------------------------
__global__ void init_nodes(float* m, float* den, int n) {
    int i = blockIdx.x * blockDim.x + threadIdx.x;
    if (i < n) { m[i] = -INFINITY; den[i] = 0.0f; }
}

// wave per edge: e = att . leakyrelu(xl[src]+xr[dst]); atomic segment max
__global__ __launch_bounds__(256)
void edge_logit(const float* __restrict__ xl, const float* __restrict__ xr,
                const float* __restrict__ att,
                const int* __restrict__ src, const int* __restrict__ dst,
                float* __restrict__ e, float* __restrict__ m, int E)
{
    int ew   = blockIdx.x * 8 + (threadIdx.x >> 5);
    int lane = threadIdx.x & 31;
    if (ew >= E) return;
    int s = src[ew], d = dst[ew];
    float4 a  = ld4(xl  + (size_t)s * DD + lane * 4);
    float4 b  = ld4(xr  + (size_t)d * DD + lane * 4);
    float4 at = ld4(att + lane * 4);
    float vx = a.x + b.x, vy = a.y + b.y, vz = a.z + b.z, vw = a.w + b.w;
    vx = vx > 0.0f ? vx : 0.2f * vx;
    vy = vy > 0.0f ? vy : 0.2f * vy;
    vz = vz > 0.0f ? vz : 0.2f * vz;
    vw = vw > 0.0f ? vw : 0.2f * vw;
    float sum = vx * at.x + vy * at.y + vz * at.z + vw * at.w;
    #pragma unroll
    for (int off = 16; off > 0; off >>= 1) sum += __shfl_xor(sum, off, 32);
    if (lane == 0) { e[ew] = sum; atomicMaxF(m + d, sum); }
}

// thread per edge: ex = exp(e - m[dst]); den[dst] += ex  (e overwritten with ex)
__global__ void edge_exp(const int* __restrict__ dst, const float* __restrict__ m,
                         float* __restrict__ e, float* __restrict__ den, int E)
{
    int i = blockIdx.x * blockDim.x + threadIdx.x;
    if (i < E) {
        int d = dst[i];
        float ex = __expf(e[i] - m[d]);
        e[i] = ex;
        atomicAdd(den + d, ex);
    }
}

// h[i] += bias[i % 128]   (in-place residual + bias before aggregation)
__global__ void add_bias(float* __restrict__ h, const float* __restrict__ bias, size_t nd) {
    size_t i = (size_t)blockIdx.x * blockDim.x + threadIdx.x;
    if (i < nd) h[i] += bias[i & (DD - 1)];
}

// wave per edge: h[dst] += (ex/(den[dst]+eps)) * xl[src]
__global__ __launch_bounds__(256)
void edge_agg(const float* __restrict__ xl,
              const int* __restrict__ src, const int* __restrict__ dst,
              const float* __restrict__ e, const float* __restrict__ den,
              float* __restrict__ h, int E)
{
    int ew   = blockIdx.x * 8 + (threadIdx.x >> 5);
    int lane = threadIdx.x & 31;
    if (ew >= E) return;
    int s = src[ew], d = dst[ew];
    float alpha = e[ew] / (den[d] + 1e-16f);
    float4 v = ld4(xl + (size_t)s * DD + lane * 4);
    float* hp = h + (size_t)d * DD + lane * 4;
    atomicAdd(hp + 0, alpha * v.x);
    atomicAdd(hp + 1, alpha * v.y);
    atomicAdd(hp + 2, alpha * v.z);
    atomicAdd(hp + 3, alpha * v.w);
}

__global__ void relu_k(float* __restrict__ h, size_t nd) {
    size_t i = (size_t)blockIdx.x * blockDim.x + threadIdx.x;
    if (i < nd) { float v = h[i]; h[i] = v > 0.0f ? v : 0.0f; }
}

// ---------------------------------------------------------------------------
extern "C" void kernel_launch(void* const* d_in, const int* in_sizes, int n_in,
                              void* d_out, int out_size, void* d_ws, size_t ws_size,
                              hipStream_t stream)
{
    // Inputs (setup_inputs dict order):
    const float* x     = (const float*)d_in[0];
    const float* Wl0   = (const float*)d_in[1];
    const float* bl0   = (const float*)d_in[2];
    const float* Wr0   = (const float*)d_in[3];
    const float* br0   = (const float*)d_in[4];
    const float* att0  = (const float*)d_in[5];
    const float* Wres0 = (const float*)d_in[6];
    const float* bias0 = (const float*)d_in[7];
    const float* Wl1   = (const float*)d_in[8];
    const float* bl1   = (const float*)d_in[9];
    const float* Wr1   = (const float*)d_in[10];
    const float* br1   = (const float*)d_in[11];
    const float* att1  = (const float*)d_in[12];
    const float* Wres1 = (const float*)d_in[13];
    const float* bias1 = (const float*)d_in[14];
    const int*   ei    = (const int*)d_in[15];

    const int NT = in_sizes[0] / DD;       // 200000 nodes
    const int E  = in_sizes[15] / 2;       // 600000 edges
    const int* src = ei;
    const int* dst = ei + E;

    const size_t ND = (size_t)NT * DD;
    float* ws   = (float*)d_ws;
    float* xl   = ws;                // [N,D]
    float* xr   = xl + ND;           // [N,D]
    float* h    = xr + ND;           // [N,D]  residual+agg accumulator / layer-1 input
    float* e    = h + ND;            // [E]
    float* m    = e + E;             // [N]
    float* den  = m + NT;            // [N]
    float* out  = (float*)d_out;     // [N,D]

    const int gTiles = (NT + 127) / 128;
    const int gE8    = (E + 7) / 8;
    const int gEt    = (E + 255) / 256;
    const int gN     = (NT + 255) / 256;
    const int gND    = (int)((ND + 255) / 256);

    // ---------------- Layer 0 ----------------
    gemm3_wmma<<<gTiles, 256, 0, stream>>>(x, NT, Wl0, bl0, xl, Wr0, br0, xr, Wres0, nullptr, h);
    init_nodes<<<gN, 256, 0, stream>>>(m, den, NT);
    edge_logit<<<gE8, 256, 0, stream>>>(xl, xr, att0, src, dst, e, m, E);
    edge_exp  <<<gEt, 256, 0, stream>>>(dst, m, e, den, E);
    add_bias  <<<gND, 256, 0, stream>>>(h, bias0, ND);      // h = x@Wres + bias
    edge_agg  <<<gE8, 256, 0, stream>>>(xl, src, dst, e, den, h, E);
    relu_k    <<<gND, 256, 0, stream>>>(h, ND);

    // ---------------- Layer 1 ----------------
    gemm3_wmma<<<gTiles, 256, 0, stream>>>(h, NT, Wl1, bl1, xl, Wr1, br1, xr, Wres1, nullptr, out);
    init_nodes<<<gN, 256, 0, stream>>>(m, den, NT);
    edge_logit<<<gE8, 256, 0, stream>>>(xl, xr, att1, src, dst, e, m, E);
    edge_exp  <<<gEt, 256, 0, stream>>>(dst, m, e, den, E);
    add_bias  <<<gND, 256, 0, stream>>>(out, bias1, ND);    // out = h@Wres + bias
    edge_agg  <<<gE8, 256, 0, stream>>>(xl, src, dst, e, den, out, E);
    // output is h[:n_users] ++ h[n_users:] == full [N,D] buffer, already in d_out
}
